// CompactKroneckerFusion_81492709474483
// MI455X (gfx1250) — matrix operand order repaired and checked by reading, compiled, not run on hardware
//
#include <hip/hip_runtime.h>

// MI455X / gfx1250, wave32. Matrix ops via v_wmma_f32_16x16x32_bf16.

typedef __attribute__((ext_vector_type(16))) __bf16 v16bf;
typedef __attribute__((ext_vector_type(8)))  __bf16 v8bf;
typedef __attribute__((ext_vector_type(8)))  float  v8f;

#define BROWS  8192
#define DIM    1024
#define SKETCH 8192
#define OUTD   512
#define LN_EPS 1e-5f

// ---------------------------------------------------------------------------
// K1a: extract count-sketch structure: each row of S has exactly one nonzero.
// One 256-thread block per row.
// ---------------------------------------------------------------------------
__global__ __launch_bounds__(256) void extract_sketch_kernel(
    const float* __restrict__ S, int* __restrict__ idx, float* __restrict__ sgn) {
  const int row = blockIdx.x;
  const float* r = S + (size_t)row * SKETCH;
  for (int c = threadIdx.x; c < SKETCH; c += 256) {
    float v = r[c];
    if (v != 0.0f) { idx[row] = c; sgn[row] = v; }  // exactly one lane writes
  }
}

// ---------------------------------------------------------------------------
// K1b: W [SKETCH x OUTD] f32  ->  W_T [OUTD x SKETCH] bf16 (tiled transpose)
// ---------------------------------------------------------------------------
__global__ void transpose_w_kernel(const float* __restrict__ W,
                                   __bf16* __restrict__ WT) {
  __shared__ __bf16 tile[32][33];
  const int k0 = blockIdx.x * 32;
  const int n0 = blockIdx.y * 32;
  const int tx = threadIdx.x, ty = threadIdx.y;  // 32 x 8
#pragma unroll
  for (int i = 0; i < 4; ++i)
    tile[ty + 8 * i][tx] = (__bf16)W[(size_t)(k0 + ty + 8 * i) * OUTD + n0 + tx];
  __syncthreads();
#pragma unroll
  for (int i = 0; i < 4; ++i)
    WT[(size_t)(n0 + ty + 8 * i) * SKETCH + k0 + tx] = tile[tx][ty + 8 * i];
}

// ---------------------------------------------------------------------------
// K2: per batch row, scatter x1/x2 into LDS sketch accumulators (ds_add_f32),
// then ck[b,k] = s1[k]*s2[k] stored as bf16 (vectorized x4). One block/row.
// ---------------------------------------------------------------------------
__global__ __launch_bounds__(256) void sketch_ck_kernel(
    const float* __restrict__ x1, const float* __restrict__ x2,
    const int* __restrict__ idx1, const float* __restrict__ sgn1,
    const int* __restrict__ idx2, const float* __restrict__ sgn2,
    __bf16* __restrict__ ck) {
  __shared__ float s1[SKETCH];
  __shared__ float s2[SKETCH];
  const int row = blockIdx.x;
  for (int k = threadIdx.x; k < SKETCH; k += 256) { s1[k] = 0.f; s2[k] = 0.f; }
  __syncthreads();
  const float* xr1 = x1 + (size_t)row * DIM;
  const float* xr2 = x2 + (size_t)row * DIM;
  for (int d = threadIdx.x; d < DIM; d += 256) {
    atomicAdd(&s1[idx1[d]], sgn1[d] * xr1[d]);
    atomicAdd(&s2[idx2[d]], sgn2[d] * xr2[d]);
  }
  __syncthreads();
  __bf16* o = ck + (size_t)row * SKETCH;
  typedef __attribute__((ext_vector_type(4))) __bf16 v4bf;
  for (int k = threadIdx.x * 4; k < SKETCH; k += 1024) {
    v4bf v;
#pragma unroll
    for (int i = 0; i < 4; ++i) v[i] = (__bf16)(s1[k + i] * s2[k + i]);
    *reinterpret_cast<v4bf*>(o + k) = v;
  }
}

// ---------------------------------------------------------------------------
// K3: h = ck @ W + b, LayerNorm, ReLU.  Block = 512 threads (16 waves),
// output tile 64 rows x 512 cols (full OUT width so LN fuses in-block).
// WMMA bf16 16x16x32, fp32 accumulate. Double-buffered LDS staging; the
// stage buffers alias the h epilogue buffer (131072 B dynamic LDS).
// ---------------------------------------------------------------------------
__global__ __launch_bounds__(512) void gemm_ln_relu_kernel(
    const __bf16* __restrict__ ck, const __bf16* __restrict__ WT,
    const float* __restrict__ bias, const float* __restrict__ gamma,
    const float* __restrict__ beta, float* __restrict__ out) {
  extern __shared__ char smem[];
  __bf16* sA = (__bf16*)smem;             // 2 x (64 x 40) bf16 (rows padded)
  __bf16* sB = (__bf16*)(smem + 10240);   // 2 x (512 x 40) bf16
  float*  sH = (float*)smem;              // 64 x 512 f32 (aliases stages)
  const int ASTRIDE = 64 * 40;            // elements per A buffer
  const int BSTRIDE = 512 * 40;           // elements per B buffer

  const int tid  = threadIdx.x;
  const int lane = tid & 31;
  const int wave = tid >> 5;
  const int row0 = blockIdx.x * 64;

  const int tm = wave & 3;   // M-tile (16 rows each)
  const int ng = wave >> 2;  // N-group of 8 tiles (128 cols)

  // ISA 16-bit A/B fragment layout: lane<16 -> K{0..7,16..23}, else K{8..15,24..31}
  const int mrow = tm * 16 + (lane & 15);
  const int koff = (lane >> 4) * 8;

  v8f acc[8] = {};

  const int a_r = tid >> 3;        // 0..63
  const int a_c = (tid & 7) * 4;   // 0,4,...,28
  const size_t aBase = (size_t)(row0 + a_r) * SKETCH + a_c;
  const size_t bBase = (size_t)tid * SKETCH;

  // ---- prologue: stage tile 0 into buffer 0
  {
    uint2 a0  = *reinterpret_cast<const uint2*>(ck + aBase);
    uint4 w0  = reinterpret_cast<const uint4*>(WT + bBase)[0];
    uint4 w1  = reinterpret_cast<const uint4*>(WT + bBase)[1];
    uint4 w2  = reinterpret_cast<const uint4*>(WT + bBase)[2];
    uint4 w3  = reinterpret_cast<const uint4*>(WT + bBase)[3];
    *reinterpret_cast<uint2*>(&sA[a_r * 40 + a_c])  = a0;
    *reinterpret_cast<uint4*>(&sB[tid * 40])        = w0;
    *reinterpret_cast<uint4*>(&sB[tid * 40 + 8])    = w1;
    *reinterpret_cast<uint4*>(&sB[tid * 40 + 16])   = w2;
    *reinterpret_cast<uint4*>(&sB[tid * 40 + 24])   = w3;
  }
  __syncthreads();

  for (int kc = 0; kc < SKETCH; kc += 32) {
    const int cur = (kc >> 5) & 1;
    const bool hasNext = (kc + 32) < SKETCH;

    // issue next tile's global loads early (latency hidden behind WMMAs)
    uint2 aN = {};
    uint4 bN0 = {}, bN1 = {}, bN2 = {}, bN3 = {};
    if (hasNext) {
      aN  = *reinterpret_cast<const uint2*>(ck + aBase + kc + 32);
      bN0 = reinterpret_cast<const uint4*>(WT + bBase + kc + 32)[0];
      bN1 = reinterpret_cast<const uint4*>(WT + bBase + kc + 32)[1];
      bN2 = reinterpret_cast<const uint4*>(WT + bBase + kc + 32)[2];
      bN3 = reinterpret_cast<const uint4*>(WT + bBase + kc + 32)[3];
    }

    // ---- compute from current buffer
    const __bf16* cA = sA + cur * ASTRIDE;
    const __bf16* cB = sB + cur * BSTRIDE;

    v8bf alo = *reinterpret_cast<const v8bf*>(&cA[mrow * 40 + koff]);
    v8bf ahi = *reinterpret_cast<const v8bf*>(&cA[mrow * 40 + koff + 16]);
    v16bf a  = __builtin_shufflevector(alo, ahi, 0, 1, 2, 3, 4, 5, 6, 7, 8, 9,
                                       10, 11, 12, 13, 14, 15);

    // load ALL B fragments first so the 16 ds_load_b128 batch, then the 8
    // independent WMMAs can issue back-to-back (no per-WMMA dscnt stall)
    v16bf bm[8];
#pragma unroll
    for (int j = 0; j < 8; ++j) {
      const int ncol = (ng * 8 + j) * 16 + (lane & 15);
      v8bf blo = *reinterpret_cast<const v8bf*>(&cB[ncol * 40 + koff]);
      v8bf bhi = *reinterpret_cast<const v8bf*>(&cB[ncol * 40 + koff + 16]);
      bm[j] = __builtin_shufflevector(blo, bhi, 0, 1, 2, 3, 4, 5, 6, 7, 8, 9,
                                      10, 11, 12, 13, 14, 15);
    }
#pragma unroll
    for (int j = 0; j < 8; ++j)
      acc[j] = __builtin_amdgcn_wmma_f32_16x16x32_bf16(
          false, a, false, bm[j], (short)0, acc[j], false, false);

    // ---- store staged registers into the other buffer
    if (hasNext) {
      __bf16* nA = sA + (cur ^ 1) * ASTRIDE;
      __bf16* nB = sB + (cur ^ 1) * BSTRIDE;
      *reinterpret_cast<uint2*>(&nA[a_r * 40 + a_c]) = aN;
      *reinterpret_cast<uint4*>(&nB[tid * 40])       = bN0;
      *reinterpret_cast<uint4*>(&nB[tid * 40 + 8])   = bN1;
      *reinterpret_cast<uint4*>(&nB[tid * 40 + 16])  = bN2;
      *reinterpret_cast<uint4*>(&nB[tid * 40 + 24])  = bN3;
    }
    __syncthreads();
  }

  // spill accumulators to LDS h tile (C layout: VGPR g -> M = g + (lane<16?0:8))
  const int rbase = tm * 16 + ((lane < 16) ? 0 : 8);
  const int cbase = lane & 15;
#pragma unroll
  for (int j = 0; j < 8; ++j) {
    const int col = (ng * 8 + j) * 16 + cbase;
#pragma unroll
    for (int g = 0; g < 8; ++g) sH[(rbase + g) * OUTD + col] = acc[j][g];
  }
  __syncthreads();

  // fused bias + LayerNorm + ReLU: 4 rows per wave, wave32 shfl reduction
  for (int rr = 0; rr < 4; ++rr) {
    const int r = wave * 4 + rr;
    float sum = 0.f, sq = 0.f;
    for (int c = lane; c < OUTD; c += 32) {
      float v = sH[r * OUTD + c] + bias[c];
      sum += v;
      sq += v * v;
    }
    for (int off = 16; off > 0; off >>= 1) {
      sum += __shfl_xor(sum, off, 32);
      sq  += __shfl_xor(sq, off, 32);
    }
    const float mu  = sum * (1.0f / OUTD);
    const float var = sq * (1.0f / OUTD) - mu * mu;
    const float rs  = rsqrtf(var + LN_EPS);
    float* orow = out + (size_t)(row0 + r) * OUTD;
    for (int c = lane; c < OUTD; c += 32) {
      float v = sH[r * OUTD + c] + bias[c];
      float o = (v - mu) * rs * gamma[c] + beta[c];
      orow[c] = fmaxf(o, 0.0f);
    }
  }
}

// ---------------------------------------------------------------------------
extern "C" void kernel_launch(void* const* d_in, const int* in_sizes, int n_in,
                              void* d_out, int out_size, void* d_ws,
                              size_t ws_size, hipStream_t stream) {
  const float* x1    = (const float*)d_in[0];
  const float* x2    = (const float*)d_in[1];
  const float* S1    = (const float*)d_in[2];
  const float* S2    = (const float*)d_in[3];
  const float* W     = (const float*)d_in[4];
  const float* b     = (const float*)d_in[5];
  const float* gamma = (const float*)d_in[6];
  const float* beta  = (const float*)d_in[7];
  float* out = (float*)d_out;

  // workspace layout (bytes):
  //   ck   bf16 [8192 x 8192]  @ 0          (134217728)
  //   W_T  bf16 [512 x 8192]   @ 134217728  (8388608)
  //   idx1/sgn1/idx2/sgn2      @ 142606336  (4 x 4096)
  char* ws = (char*)d_ws;
  __bf16* ck  = (__bf16*)ws;
  __bf16* WT  = (__bf16*)(ws + 134217728);
  int*   idx1 = (int*)  (ws + 142606336);
  float* sgn1 = (float*)(ws + 142610432);
  int*   idx2 = (int*)  (ws + 142614528);
  float* sgn2 = (float*)(ws + 142618624);

  extract_sketch_kernel<<<DIM, 256, 0, stream>>>(S1, idx1, sgn1);
  extract_sketch_kernel<<<DIM, 256, 0, stream>>>(S2, idx2, sgn2);
  transpose_w_kernel<<<dim3(SKETCH / 32, OUTD / 32), dim3(32, 8), 0, stream>>>(W, WT);
  sketch_ck_kernel<<<BROWS, 256, 0, stream>>>(x1, x2, idx1, sgn1, idx2, sgn2, ck);
  gemm_ln_relu_kernel<<<BROWS / 64, 512, 131072, stream>>>(ck, WT, b, gamma, beta, out);
}
